// GNNValue_37237366456827
// MI455X (gfx1250) — compile-verified, matrix operand order checked by reading, not run on hardware
//
#include <hip/hip_runtime.h>
#include <hip/hip_bf16.h>

// Problem constants (match reference)
#define Nn 50000
#define Ee 800000
#define ETOT 850000   // edges + self loops
#define Gg 8
#define Hdim 64

typedef __attribute__((ext_vector_type(2)))  float    v2f;
typedef __attribute__((ext_vector_type(8)))  float    v8f;
typedef __attribute__((ext_vector_type(16))) _Float16 v16h;

#if defined(__has_builtin)
#if __has_builtin(__builtin_amdgcn_wmma_f32_16x16x4_f32)
#define GAT_USE_WMMA_F32X4 1
#endif
#endif

// ---- order-preserving float<->u32 map for atomicMax on signed floats ----
__device__ __forceinline__ unsigned flipF(float f) {
  unsigned u = __float_as_uint(f);
  return (u & 0x80000000u) ? ~u : (u | 0x80000000u);
}
__device__ __forceinline__ float unflipF(unsigned u) {
  u = (u & 0x80000000u) ? (u & 0x7FFFFFFFu) : ~u;
  return __uint_as_float(u);
}

__global__ void zero32_kernel(unsigned* __restrict__ p, long n) {
  long i = (long)blockIdx.x * blockDim.x + threadIdx.x;
  if (i < n) p[i] = 0u;
}

// Layer-1 features: h = x @ W1 (K=3), e_s = h.a_src, e_d = h.a_dst.
// One wave32 per node, 2 features per lane, shuffle reduce for the dots.
__global__ void feat1_kernel(const float* __restrict__ x,  const float* __restrict__ W1,
                             const float* __restrict__ as, const float* __restrict__ ad,
                             float* __restrict__ Hout, float* __restrict__ es, float* __restrict__ ed) {
  int node = blockIdx.x * 8 + (threadIdx.x >> 5);
  int lane = threadIdx.x & 31;
  if (node >= Nn) return;
  float x0 = x[node * 3 + 0], x1 = x[node * 3 + 1], x2 = x[node * 3 + 2];
  float ss = 0.f, sd = 0.f;
#pragma unroll
  for (int j = 0; j < 2; ++j) {
    int c = lane + 32 * j;
    float h = fmaf(x0, W1[c], fmaf(x1, W1[64 + c], x2 * W1[128 + c]));
    Hout[(size_t)node * Hdim + c] = h;
    ss = fmaf(h, as[c], ss);
    sd = fmaf(h, ad[c], sd);
  }
#pragma unroll
  for (int off = 16; off > 0; off >>= 1) {
    ss += __shfl_xor(ss, off);
    sd += __shfl_xor(sd, off);
  }
  if (lane == 0) { es[node] = ss; ed[node] = sd; }
}

// Per-node attention logits for layer 2 (h already computed)
__global__ void esd_kernel(const float* __restrict__ Hin, const float* __restrict__ as,
                           const float* __restrict__ ad, float* __restrict__ es, float* __restrict__ ed) {
  int node = blockIdx.x * 8 + (threadIdx.x >> 5);
  int lane = threadIdx.x & 31;
  if (node >= Nn) return;
  float ss = 0.f, sd = 0.f;
#pragma unroll
  for (int j = 0; j < 2; ++j) {
    int c = lane + 32 * j;
    float h = Hin[(size_t)node * Hdim + c];
    ss = fmaf(h, as[c], ss);
    sd = fmaf(h, ad[c], sd);
  }
#pragma unroll
  for (int off = 16; off > 0; off >>= 1) {
    ss += __shfl_xor(ss, off);
    sd += __shfl_xor(sd, off);
  }
  if (lane == 0) { es[node] = ss; ed[node] = sd; }
}

// Edge pass A: segment max of leaky_relu(e_s[src]+e_d[dst]) over dst
__global__ void edge_max_kernel(const int* __restrict__ ei, const float* __restrict__ es,
                                const float* __restrict__ ed, unsigned* __restrict__ m_u) {
  int e = blockIdx.x * 256 + threadIdx.x;
  if (e >= ETOT) return;
  int s, d;
  if (e < Ee) { s = ei[e]; d = ei[Ee + e]; } else { s = d = e - Ee; }
  float lg = es[s] + ed[d];
  lg = lg > 0.f ? lg : 0.2f * lg;
  atomicMax(&m_u[d], flipF(lg));
}

// Edge pass B (fused): agg[dst] += exp(e - m[dst]) * h[src]; denom[dst] += exp(...)
// 64 threads per edge (one per feature); coalesced row gather + row scatter-add.
__global__ void edge_acc_kernel(const int* __restrict__ ei, const float* __restrict__ es,
                                const float* __restrict__ ed, const unsigned* __restrict__ m_u,
                                const float* __restrict__ Hin, float* __restrict__ agg,
                                float* __restrict__ denom) {
  long gid = (long)blockIdx.x * 256 + threadIdx.x;
  long e = gid >> 6;
  int c = (int)(gid & 63);
  if (e >= ETOT) return;
  int s, d;
  if (e < Ee) { s = ei[e]; d = ei[Ee + e]; } else { s = d = (int)(e - Ee); }
  float lg = es[s] + ed[d];
  lg = lg > 0.f ? lg : 0.2f * lg;
  float ex = __expf(lg - unflipF(m_u[d]));
  if (c == 0) atomicAdd(&denom[d], ex);
  atomicAdd(&agg[(size_t)d * Hdim + c], ex * Hin[(size_t)s * Hdim + c]);
}

// In-place: agg = relu(agg/denom + b)
__global__ void finalize_kernel(float* __restrict__ agg, const float* __restrict__ denom,
                                const float* __restrict__ b) {
  long gid = (long)blockIdx.x * 256 + threadIdx.x;
  if (gid >= (long)Nn * Hdim) return;
  int i = (int)(gid >> 6), c = (int)(gid & 63);
  float v = agg[gid] / denom[i] + b[c];
  agg[gid] = v > 0.f ? v : 0.f;
}

// Same + fused per-graph mean-pool accumulation
__global__ void finalize_pool_kernel(const float* __restrict__ agg, const float* __restrict__ denom,
                                     const float* __restrict__ b, const int* __restrict__ batch,
                                     float* __restrict__ pool, float* __restrict__ cnt) {
  long gid = (long)blockIdx.x * 256 + threadIdx.x;
  if (gid >= (long)Nn * Hdim) return;
  int i = (int)(gid >> 6), c = (int)(gid & 63);
  float v = agg[gid] / denom[i] + b[c];
  v = v > 0.f ? v : 0.f;
  int g = batch[i];
  atomicAdd(&pool[g * Hdim + c], v);
  if (c == 0) atomicAdd(&cnt[g], 1.0f);
}

// h2 = h @ W2 : [N,64] x [64,64], f32.  One wave per 16-row tile; 4 col tiles.
// Primary path: V_WMMA_F32_16X16X4_F32 (exact f32). Fallback: f16 inputs / f32 acc.
__global__ __launch_bounds__(256) void gemm64_wmma_kernel(const float* __restrict__ A,
                                                          const float* __restrict__ W,
                                                          float* __restrict__ D) {
  int wave = threadIdx.x >> 5;
  int lane = threadIdx.x & 31;
  int tile = blockIdx.x * 8 + wave;
  if (tile >= Nn / 16) return;          // wave-uniform: EXEC stays all-ones for WMMA
  int base = tile * 16;
  int mrow = lane & 15;                 // M (for A) / N (for B,D) within tile
  int hi = lane >> 4;                   // lane-half selector
  const float* Arow = A + (size_t)(base + mrow) * Hdim;

  v8f acc[4] = {};

#ifdef GAT_USE_WMMA_F32X4
  // A 16x4 f32 frag: lanes0-15 -> K={0,1}, lanes16-31 -> K={2,3} (VGPR0/1)
  // B 4x16 f32 frag: same K split, N = lane&15
#pragma unroll
  for (int kk = 0; kk < 16; ++kk) {
    int k0 = kk * 4 + hi * 2;
    v2f a;
    a.x = Arow[k0];
    a.y = Arow[k0 + 1];
#pragma unroll
    for (int nt = 0; nt < 4; ++nt) {
      int col = nt * 16 + mrow;
      v2f bf;
      bf.x = W[(size_t)k0 * Hdim + col];
      bf.y = W[(size_t)(k0 + 1) * Hdim + col];
      acc[nt] = __builtin_amdgcn_wmma_f32_16x16x4_f32(
          false, a, false, bf, (short)0, acc[nt], false, false);
    }
  }
#else
  // Fallback: f32->f16 convert, K=64 as two 16x16x32 f16 WMMAs per col tile.
  // 16-bit A 16x32 layout: VGPR r holds K = (r&3)*2+{0,1} + (r>=4?16:0) + (hi?8:0)
#pragma unroll
  for (int t = 0; t < 2; ++t) {
    v16h a;
#pragma unroll
    for (int r = 0; r < 8; ++r) {
#pragma unroll
      for (int e2 = 0; e2 < 2; ++e2) {
        int K = t * 32 + (r & 3) * 2 + e2 + ((r >> 2) * 16) + hi * 8;
        a[2 * r + e2] = (_Float16)Arow[K];
      }
    }
#pragma unroll
    for (int nt = 0; nt < 4; ++nt) {
      int col = nt * 16 + mrow;
      v16h bf;
#pragma unroll
      for (int r = 0; r < 8; ++r) {
#pragma unroll
        for (int e2 = 0; e2 < 2; ++e2) {
          int K = t * 32 + (r & 3) * 2 + e2 + ((r >> 2) * 16) + hi * 8;
          bf[2 * r + e2] = (_Float16)W[(size_t)K * Hdim + col];
        }
      }
      acc[nt] = __builtin_amdgcn_wmma_f32_16x16x32_f16(
          false, a, false, bf, (short)0, acc[nt], false, false);
    }
  }
#endif

  // D 16x16 f32 layout: VGPR r -> M = r + (hi?8:0), N = lane&15
#pragma unroll
  for (int nt = 0; nt < 4; ++nt) {
#pragma unroll
    for (int r = 0; r < 8; ++r) {
      D[(size_t)(base + r + hi * 8) * Hdim + nt * 16 + mrow] = acc[nt][r];
    }
  }
}

// Head: pooled = sums/max(cnt,1); out = relu(pooled@Wh1+bh1)@Wh2+bh2. One block.
__global__ void head_kernel(const float* __restrict__ pool, const float* __restrict__ cnt,
                            const float* __restrict__ Wh1, const float* __restrict__ bh1,
                            const float* __restrict__ Wh2, const float* __restrict__ bh2,
                            float* __restrict__ out) {
  __shared__ float z[Gg * 128];
  int k = threadIdx.x;  // 0..127
  for (int g = 0; g < Gg; ++g) {
    float c = cnt[g];
    float inv = 1.0f / fmaxf(c, 1.0f);
    float acc = bh1[k];
    for (int cc = 0; cc < Hdim; ++cc)
      acc = fmaf(pool[g * Hdim + cc] * inv, Wh1[cc * 128 + k], acc);
    z[g * 128 + k] = fmaxf(acc, 0.f);
  }
  __syncthreads();
  if (k < Gg) {
    float acc = bh2[0];
    for (int j = 0; j < 128; ++j) acc = fmaf(z[k * 128 + j], Wh2[j], acc);
    out[k] = acc;
  }
}

extern "C" void kernel_launch(void* const* d_in, const int* in_sizes, int n_in,
                              void* d_out, int out_size, void* d_ws, size_t ws_size,
                              hipStream_t stream) {
  const float* x      = (const float*)d_in[0];
  const int*   ei     = (const int*)d_in[1];
  const int*   batch  = (const int*)d_in[2];
  const float* W1     = (const float*)d_in[3];
  const float* a_src1 = (const float*)d_in[4];
  const float* a_dst1 = (const float*)d_in[5];
  const float* b1     = (const float*)d_in[6];
  const float* W2     = (const float*)d_in[7];
  const float* a_src2 = (const float*)d_in[8];
  const float* a_dst2 = (const float*)d_in[9];
  const float* b2     = (const float*)d_in[10];
  const float* Wh1    = (const float*)d_in[11];
  const float* bh1    = (const float*)d_in[12];
  const float* Wh2    = (const float*)d_in[13];
  const float* bh2    = (const float*)d_in[14];
  float* out = (float*)d_out;

  // workspace layout (floats)
  float* ws    = (float*)d_ws;
  float* bufA  = ws;                          // N*64
  float* bufB  = bufA + (size_t)Nn * Hdim;    // N*64
  float* es    = bufB + (size_t)Nn * Hdim;    // N
  float* ed    = es + Nn;                     // N
  unsigned* m_u = (unsigned*)(ed + Nn);       // N
  float* denom = ed + 2 * (size_t)Nn;         // N (right after m_u)
  float* pool  = denom + Nn;                  // G*64
  float* cnt   = pool + Gg * Hdim;            // G

  const long NH = (long)Nn * Hdim;
  dim3 b256(256);
  int gZeroNH  = (int)((NH + 255) / 256);
  int gZeroAux = (int)((2L * Nn + 255) / 256);
  int gNode    = (Nn + 7) / 8;
  int gEdge    = (ETOT + 255) / 256;
  int gEdgeF   = (int)(((long)ETOT * 64 + 255) / 256);
  int gNF      = (int)((NH + 255) / 256);
  int gGemm    = ((Nn / 16) + 7) / 8;

  // ---------------- Layer 1 ----------------
  feat1_kernel<<<gNode, b256, 0, stream>>>(x, W1, a_src1, a_dst1, bufA, es, ed);
  zero32_kernel<<<gZeroNH, b256, 0, stream>>>((unsigned*)bufB, NH);
  zero32_kernel<<<gZeroAux, b256, 0, stream>>>(m_u, 2L * Nn);          // m_u + denom
  zero32_kernel<<<3, b256, 0, stream>>>((unsigned*)pool, Gg * Hdim + Gg); // pool + cnt
  edge_max_kernel<<<gEdge, b256, 0, stream>>>(ei, es, ed, m_u);
  edge_acc_kernel<<<gEdgeF, b256, 0, stream>>>(ei, es, ed, m_u, bufA, bufB, denom);
  finalize_kernel<<<gNF, b256, 0, stream>>>(bufB, denom, b1);          // bufB = h (layer-2 input)

  // ---------------- Layer 2 ----------------
  gemm64_wmma_kernel<<<gGemm, b256, 0, stream>>>(bufB, W2, bufA);      // bufA = h @ W2 (WMMA)
  esd_kernel<<<gNode, b256, 0, stream>>>(bufA, a_src2, a_dst2, es, ed);
  zero32_kernel<<<gZeroNH, b256, 0, stream>>>((unsigned*)bufB, NH);
  zero32_kernel<<<gZeroAux, b256, 0, stream>>>(m_u, 2L * Nn);
  edge_max_kernel<<<gEdge, b256, 0, stream>>>(ei, es, ed, m_u);
  edge_acc_kernel<<<gEdgeF, b256, 0, stream>>>(ei, es, ed, m_u, bufA, bufB, denom);
  finalize_pool_kernel<<<gNF, b256, 0, stream>>>(bufB, denom, b2, batch, pool, cnt);

  // ---------------- Head ----------------
  head_kernel<<<1, 128, 0, stream>>>(pool, cnt, Wh1, bh1, Wh2, bh2, out);
  (void)in_sizes; (void)n_in; (void)out_size; (void)ws_size;
}
